// DGraphCast_26585847562368
// MI455X (gfx1250) — matrix operand order repaired and checked by reading, not compiled, not run on hardware
//
#include <hip/hip_runtime.h>
#include <hip/hip_bf16.h>
#include <cstdint>
#include <cstddef>

typedef __attribute__((ext_vector_type(16))) _Float16 v16h;
typedef __attribute__((ext_vector_type(8)))  _Float16 v8h;
typedef __attribute__((ext_vector_type(4)))  _Float16 v4h;
typedef __attribute__((ext_vector_type(8)))  float    v8f;
typedef __attribute__((ext_vector_type(4)))  float    v4f32;  // plain vector (AS-safe)
typedef int vsi4 __attribute__((vector_size(16)));  // matches async-LDS builtin param

#define AS1 __attribute__((address_space(1)))
#define AS3 __attribute__((address_space(3)))
#define HD __device__ __forceinline__

#if defined(__has_builtin)
#if __has_builtin(__builtin_amdgcn_global_load_async_to_lds_b128) && \
    __has_builtin(__builtin_amdgcn_s_wait_asynccnt)
#define HAVE_ASYNC_LDS 1
#endif
#endif

// ---------------------------------------------------------------------------
// WMMA fragment helpers (wave32, V_WMMA_F32_16X16X32_F16)
// A 16x32 f16: lane l (m=l&15, half=l>>4) holds K = k0+half*8..+7 and
//              K = k0+16+half*8..+7  (two contiguous 16B runs per row).
// B 32x16 f16: lane l holds K = k0+half*8..+7 and k0+16+half*8..+7 for N = nt*16+(l&15);
//              weights pre-packed so each lane reads 32 contiguous bytes.
// C/D 16x16 f32: VGPR v, lane l -> row = v + 8*(l>>4), col = (l&15) + 16*nt.
// All LDS pointers are kept in addrspace(3): 32-bit addressing, no per-access
// generic->LDS casts (they were generating cmp/cndmask chains).
// ---------------------------------------------------------------------------

HD v16h mk16(v8h lo, v8h hi) {
  v16h r;
#pragma unroll
  for (int j = 0; j < 8; ++j) { r[j] = lo[j]; r[j + 8] = hi[j]; }
  return r;
}

// A-fragment from global f16 row
HD v16h afrag_g16(const AS1 _Float16* rowp, int k0, int lane) {
  const int h8 = ((lane >> 4) & 1) * 8;
  v8h lo = *(const AS1 v8h*)(rowp + k0 + h8);
  v8h hi = *(const AS1 v8h*)(rowp + k0 + 16 + h8);
  return mk16(lo, hi);
}

// A-fragment from global f32 row (segment-sum accumulator)
HD v16h afrag_g32(const AS1 float* rowp, int k0, int lane) {
  const int h8 = ((lane >> 4) & 1) * 8;
  v4f32 a0 = *(const AS1 v4f32*)(rowp + k0 + h8);
  v4f32 a1 = *(const AS1 v4f32*)(rowp + k0 + h8 + 4);
  v4f32 b0 = *(const AS1 v4f32*)(rowp + k0 + 16 + h8);
  v4f32 b1 = *(const AS1 v4f32*)(rowp + k0 + 16 + h8 + 4);
  v16h r;
#pragma unroll
  for (int j = 0; j < 4; ++j) {
    r[j]      = (_Float16)a0[j];
    r[j + 4]  = (_Float16)a1[j];
    r[j + 8]  = (_Float16)b0[j];
    r[j + 12] = (_Float16)b1[j];
  }
  return r;
}

// A-fragment from LDS f16 row (addrspace(3) throughout)
HD v16h afrag_l16(const AS3 _Float16* rowp, int k0, int lane) {
  const int h8 = ((lane >> 4) & 1) * 8;
  v8h lo = *(const AS3 v8h*)(rowp + k0 + h8);
  v8h hi = *(const AS3 v8h*)(rowp + k0 + 16 + h8);
  return mk16(lo, hi);
}

// B-fragment from LDS-staged packed weights (addrspace(3))
HD v16h bfrag_l(const AS3 _Float16* wp, int tile, int lane) {
  const AS3 _Float16* p = wp + (tile * 32 + lane) * 16;
  v8h lo = *(const AS3 v8h*)p;
  v8h hi = *(const AS3 v8h*)(p + 8);
  return mk16(lo, hi);
}

HD float silu(float x) { return x / (1.f + __expf(-x)); }

// Cooperative stage of packed weights (global f16 -> LDS), async when available
HD void stage(const _Float16* g, AS3 _Float16* l, int halfs) {
  const int tid = threadIdx.x;
#if HAVE_ASYNC_LDS
  const AS1 char* gs = (const AS1 char*)(const AS1 _Float16*)g;
  AS3 char* ls = (AS3 char*)l;
  const int bytes = halfs * 2;
  for (int off = tid * 16; off < bytes; off += 256 * 16)
    __builtin_amdgcn_global_load_async_to_lds_b128(
        (AS1 vsi4*)(gs + off), (AS3 vsi4*)(ls + off), 0, 0);
#else
  const AS1 v8h* gs = (const AS1 v8h*)(const AS1 _Float16*)g;
  for (int off8 = tid; off8 * 8 < halfs; off8 += 256)
    *(AS3 v8h*)(l + off8 * 8) = gs[off8];
#endif
}

HD void stage_wait() {
#if HAVE_ASYNC_LDS
  __builtin_amdgcn_s_wait_asynccnt(0);
#endif
  __syncthreads();
}

// ---------------------------------------------------------------------------
// Second layer (128 -> DOUT) from f16 hidden in LDS (Hw) with LDS weights
// (W2), + optional LayerNorm, optional f16 residual; stores f16 or f32.
// ---------------------------------------------------------------------------
template <int DOUT, bool OF32>
__device__ void layer2_ln_store(const AS3 _Float16* Hw, const AS3 _Float16* W2,
                                const float* b2, const float* gam, const float* bet,
                                const _Float16* res, void* outp,
                                int rowbase, int rows, int lane) {
  constexpr int NT2 = DOUT / 16;
  const int m = lane & 15, half = lane >> 4;
  const AS3 _Float16* hrow = Hw + m * 128;

  v8f acc[NT2];
#pragma unroll
  for (int nt = 0; nt < NT2; ++nt)
#pragma unroll
    for (int v = 0; v < 8; ++v) acc[nt][v] = 0.f;

#pragma unroll
  for (int kt = 0; kt < 4; ++kt) {
    v16h a = afrag_l16(hrow, kt * 32, lane);
#pragma unroll
    for (int nt = 0; nt < NT2; ++nt) {
      v16h b = bfrag_l(W2, kt * NT2 + nt, lane);
      acc[nt] = __builtin_amdgcn_wmma_f32_16x16x32_f16(
          false, a, false, b, (short)0, acc[nt], false, false);
    }
  }

  const AS1 float* b2g = (const AS1 float*)b2;
  float bb[NT2], gg[NT2], be[NT2];
#pragma unroll
  for (int nt = 0; nt < NT2; ++nt) {
    int c = nt * 16 + m;
    bb[nt] = b2g[c];
    if (gam) { gg[nt] = ((const AS1 float*)gam)[c]; be[nt] = ((const AS1 float*)bet)[c]; }
  }
#pragma unroll
  for (int nt = 0; nt < NT2; ++nt)
#pragma unroll
    for (int v = 0; v < 8; ++v) acc[nt][v] += bb[nt];

  float mu[8], rs[8];
  if (gam) {
#pragma unroll
    for (int v = 0; v < 8; ++v) {
      float s = 0.f, s2 = 0.f;
#pragma unroll
      for (int nt = 0; nt < NT2; ++nt) { float x = acc[nt][v]; s += x; s2 += x * x; }
      for (int mk = 1; mk < 16; mk <<= 1) {
        s  += __shfl_xor(s,  mk, 32);
        s2 += __shfl_xor(s2, mk, 32);
      }
      float mean = s * (1.f / DOUT);
      float var  = s2 * (1.f / DOUT) - mean * mean;
      mu[v] = mean;
      rs[v] = rsqrtf(var + 1e-5f);
    }
  }

#pragma unroll
  for (int v = 0; v < 8; ++v) {
    int rg = rowbase + v + 8 * half;
    if (rg >= rows) continue;
#pragma unroll
    for (int nt = 0; nt < NT2; ++nt) {
      int c = nt * 16 + m;
      float x = acc[nt][v];
      if (gam) x = (x - mu[v]) * rs[v] * gg[nt] + be[nt];
      if (res) x += (float)((const AS1 _Float16*)res)[(size_t)rg * DOUT + c];
      if (OF32) ((AS1 float*)outp)[(size_t)rg * DOUT + c] = x;
      else ((AS1 _Float16*)outp)[(size_t)rg * DOUT + c] = (_Float16)x;
    }
  }
}

// ---------------------------------------------------------------------------
// Embedder MLP: din in {3,4,64}: scalar layer1 (f32 in) -> LDS f16 hidden
// (overlapped with async weight staging) -> WMMA layer2 + LN -> f16 out.
// ---------------------------------------------------------------------------
__global__ void __launch_bounds__(256) embed_k(
    const float* __restrict__ x, int din, int rows,
    const float* __restrict__ w1, const float* __restrict__ b1,
    const _Float16* __restrict__ w2p, const float* __restrict__ b2,
    const float* __restrict__ gam, const float* __restrict__ bet,
    _Float16* __restrict__ out) {
  extern __shared__ char dynsm[];
  AS3 _Float16* Hs = (AS3 _Float16*)(AS3 char*)dynsm;  // 8 waves * 16 * 128
  AS3 _Float16* W2 = Hs + 8 * 2048;                    // 4*8 tiles
  stage(w2p, W2, 16384);

  const int lane = threadIdx.x & 31, wave = threadIdx.x >> 5;
  const int m = lane & 15, half = lane >> 4;
  const int rowbase = blockIdx.x * 128 + wave * 16;
  int rc = rowbase + m; if (rc >= rows) rc = rows - 1;
  const AS1 float* xr = (const AS1 float*)x + (size_t)rc * din;
  const AS1 float* w1g = (const AS1 float*)w1;
  const AS1 float* b1g = (const AS1 float*)b1;
  AS3 _Float16* Hw = Hs + wave * 2048;

  for (int ci = 0; ci < 64; ++ci) {
    int c = half * 64 + ci;
    float a = b1g[c];
    for (int k = 0; k < din; ++k) a += xr[k] * w1g[(size_t)k * 128 + c];
    Hw[m * 128 + c] = (_Float16)silu(a);
  }
  stage_wait();
  layer2_ln_store<128, false>(Hw, W2, b2, gam, bet, nullptr, out, rowbase, rows, lane);
}

// ---------------------------------------------------------------------------
// Fused gather-concat-MLP-LN-residual block; all weights staged to LDS.
// NSEG 128-wide f16 segments (segment 1 may be the f32 aggregation buffer).
// ---------------------------------------------------------------------------
template <int NSEG, int DOUT, bool S1F32, bool OF32>
__global__ void __launch_bounds__(256) fused_block_k(
    const void* __restrict__ s0v, const void* __restrict__ s1v,
    const void* __restrict__ s2v,
    const int* __restrict__ i1, const int* __restrict__ i2, int rows,
    const _Float16* __restrict__ w1p, const float* __restrict__ b1,
    const _Float16* __restrict__ w2p, const float* __restrict__ b2,
    const float* __restrict__ gam, const float* __restrict__ bet,
    const _Float16* __restrict__ res, void* __restrict__ outp) {
  extern __shared__ char dynsm[];
  AS3 _Float16* Hs = (AS3 _Float16*)(AS3 char*)dynsm;  // 32 KB hidden
  AS3 _Float16* W1 = Hs + 8 * 2048;                    // NSEG*32 KB
  AS3 _Float16* W2 = W1 + NSEG * 16384;                // (DOUT/16)*4 KB
  stage(w1p, W1, NSEG * 16384);
  stage(w2p, W2, 4 * (DOUT / 16) * 512);

  const int lane = threadIdx.x & 31, wave = threadIdx.x >> 5;
  const int m = lane & 15, half = lane >> 4;
  const int rowbase = blockIdx.x * 128 + wave * 16;
  int rc = rowbase + m; if (rc >= rows) rc = rows - 1;

  const AS1 _Float16* ph[3];
  const AS1 float* pf1 = nullptr;
  ph[0] = (const AS1 _Float16*)s0v + (size_t)rc * 128;
  if (NSEG > 1) {
    int r1 = i1 ? ((const AS1 int*)i1)[rc] : rc;
    if (S1F32) pf1 = (const AS1 float*)s1v + (size_t)r1 * 128;
    else ph[1] = (const AS1 _Float16*)s1v + (size_t)r1 * 128;
  }
  if (NSEG > 2) {
    int r2 = i2 ? ((const AS1 int*)i2)[rc] : rc;
    ph[2] = (const AS1 _Float16*)s2v + (size_t)r2 * 128;
  }
  stage_wait();

  v8f acc[8];
#pragma unroll
  for (int nt = 0; nt < 8; ++nt)
#pragma unroll
    for (int v = 0; v < 8; ++v) acc[nt][v] = 0.f;

#pragma unroll
  for (int kt = 0; kt < NSEG * 4; ++kt) {
    const int seg = kt >> 2, k0 = (kt & 3) * 32;
    v16h a;
    if (S1F32 && seg == 1) a = afrag_g32(pf1, k0, lane);
    else a = afrag_g16(ph[seg], k0, lane);
#pragma unroll
    for (int nt = 0; nt < 8; ++nt) {
      v16h b = bfrag_l(W1, kt * 8 + nt, lane);
      acc[nt] = __builtin_amdgcn_wmma_f32_16x16x32_f16(
          false, a, false, b, (short)0, acc[nt], false, false);
    }
  }

  const AS1 float* b1g = (const AS1 float*)b1;
  AS3 _Float16* Hw = Hs + wave * 2048;
#pragma unroll
  for (int nt = 0; nt < 8; ++nt) {
    float bb = b1g[nt * 16 + m];
#pragma unroll
    for (int v = 0; v < 8; ++v) {
      Hw[(v + 8 * half) * 128 + nt * 16 + m] = (_Float16)silu(acc[nt][v] + bb);
    }
  }
  __syncthreads();
  layer2_ln_store<DOUT, OF32>(Hw, W2, b2, gam, bet, res, outp, rowbase, rows, lane);
}

// ---------------------------------------------------------------------------
// Weight packer: f32 KxN row-major -> f16 WMMA B-fragment tiles
// ---------------------------------------------------------------------------
__global__ void pack_k(const float* __restrict__ src, _Float16* __restrict__ dst,
                       int K, int N) {
  const int NT = N / 16;
  const int tile = blockIdx.x;
  const int kt = tile / NT, nt = tile % NT;
  const int l = threadIdx.x;
  const int kb = kt * 32 + ((l >> 4) & 1) * 8;
  const int n = nt * 16 + (l & 15);
  AS1 _Float16* d = (AS1 _Float16*)dst + ((size_t)tile * 32 + l) * 16;
  const AS1 float* s = (const AS1 float*)src;
#pragma unroll
  for (int j = 0; j < 16; ++j) d[j] = (_Float16)s[(size_t)(kb + j + (j & 8)) * N + n];
}

__global__ void zero_k(float* p, size_t n) {
  size_t i = (size_t)blockIdx.x * blockDim.x + threadIdx.x;
  if (i < n) ((AS1 float*)p)[i] = 0.f;
}

// segment_sum of 128-wide f16 rows into f32 via global atomics
__global__ void scatter_add_k(const _Float16* __restrict__ e,
                              const int* __restrict__ dst,
                              float* __restrict__ agg, int E) {
  int id = blockIdx.x * blockDim.x + threadIdx.x;
  int edge = id >> 5;
  if (edge >= E) return;
  int c4 = (id & 31) * 4;
  v4h v = *(const AS1 v4h*)((const AS1 _Float16*)e + (size_t)edge * 128 + c4);
  AS1 float* a = (AS1 float*)agg +
                 (size_t)((const AS1 int*)dst)[edge] * 128 + c4;
#pragma unroll
  for (int k = 0; k < 4; ++k)
    __hip_atomic_fetch_add(a + k, (float)v[k], __ATOMIC_RELAXED,
                           __HIP_MEMORY_SCOPE_AGENT);
}

// ---------------------------------------------------------------------------
extern "C" void kernel_launch(void* const* d_in, const int* in_sizes, int n_in,
                              void* d_out, int out_size, void* d_ws, size_t ws_size,
                              hipStream_t stream) {
  (void)in_sizes; (void)n_in; (void)out_size; (void)ws_size;
  constexpr int N_GRID = 131072, N_MESH = 40962, H = 128;
  constexpr int E_M2M = 262144, E_G2M = 393216, E_M2G = 393216;

  const float* x_grid = (const float*)d_in[0];
  const float* x_mesh = (const float*)d_in[1];
  const float* x_m2m  = (const float*)d_in[2];
  const float* x_g2m  = (const float*)d_in[3];
  const float* x_m2g  = (const float*)d_in[4];
  const int* m2m_src = (const int*)d_in[5];
  const int* m2m_dst = (const int*)d_in[6];
  const int* g2m_src = (const int*)d_in[7];
  const int* g2m_dst = (const int*)d_in[8];
  const int* m2g_src = (const int*)d_in[9];
  const int* m2g_dst = (const int*)d_in[10];

  // params flattened as a JAX pytree: dict keys sorted alphabetically.
  // Leaf order within a norm-MLP: b1,b2,beta,g,w1,w2 ; final: b1,b2,w1,w2
  auto Pf = [&](int i) { return (const float*)d_in[i]; };
  struct M6 { const float *b1, *b2, *beta, *g, *w1, *w2; };
  auto m6 = [&](int b) {
    return M6{Pf(b), Pf(b + 1), Pf(b + 2), Pf(b + 3), Pf(b + 4), Pf(b + 5)};
  };
  M6 dec_edge = m6(11), dec_node = m6(17), emb_g2m = m6(23), emb_grid = m6(29),
     emb_m2g = m6(35), emb_m2m = m6(41), emb_mesh = m6(47), enc_edge = m6(53),
     enc_grid = m6(59), enc_node = m6(65);
  const float* fin_b1 = Pf(71); const float* fin_b2 = Pf(72);
  const float* fin_w1 = Pf(73); const float* fin_w2 = Pf(74);
  M6 proc_e[2] = {m6(75), m6(87)};
  M6 proc_n[2] = {m6(81), m6(93)};

  // workspace carve-out (features live in HBM as f16; AGG stays f32)
  size_t off = 0;
  auto alloc = [&](size_t bytes) -> void* {
    off = (off + 255) & ~(size_t)255;
    void* p = (char*)d_ws + off;
    off += bytes;
    return p;
  };
  _Float16* G    = (_Float16*)alloc((size_t)N_GRID * H * 2);
  _Float16* Mn   = (_Float16*)alloc((size_t)N_MESH * H * 2);
  _Float16* Em2m = (_Float16*)alloc((size_t)E_M2M * H * 2);
  _Float16* Eg2m = (_Float16*)alloc((size_t)E_G2M * H * 2);
  _Float16* Em2g = (_Float16*)alloc((size_t)E_M2G * H * 2);
  float*    AGG  = (float*)alloc((size_t)N_GRID * H * 4);  // max(N_GRID, N_MESH)

  auto pack = [&](const float* src, int K, int N) -> const _Float16* {
    _Float16* d = (_Float16*)alloc((size_t)K * N * 2);
    pack_k<<<dim3((K / 32) * (N / 16)), dim3(32), 0, stream>>>(src, d, K, N);
    return d;
  };

  const _Float16* p_eg_w2  = pack(emb_grid.w2, 128, 128);
  const _Float16* p_em_w2  = pack(emb_mesh.w2, 128, 128);
  const _Float16* p_m2m_w2 = pack(emb_m2m.w2, 128, 128);
  const _Float16* p_g2m_w2 = pack(emb_g2m.w2, 128, 128);
  const _Float16* p_m2g_w2 = pack(emb_m2g.w2, 128, 128);
  const _Float16* p_ee_w1 = pack(enc_edge.w1, 384, 128);
  const _Float16* p_ee_w2 = pack(enc_edge.w2, 128, 128);
  const _Float16* p_en_w1 = pack(enc_node.w1, 256, 128);
  const _Float16* p_en_w2 = pack(enc_node.w2, 128, 128);
  const _Float16* p_gg_w1 = pack(enc_grid.w1, 128, 128);
  const _Float16* p_gg_w2 = pack(enc_grid.w2, 128, 128);
  const _Float16 *p_pe_w1[2], *p_pe_w2[2], *p_pn_w1[2], *p_pn_w2[2];
  for (int L = 0; L < 2; ++L) {
    p_pe_w1[L] = pack(proc_e[L].w1, 384, 128);
    p_pe_w2[L] = pack(proc_e[L].w2, 128, 128);
    p_pn_w1[L] = pack(proc_n[L].w1, 256, 128);
    p_pn_w2[L] = pack(proc_n[L].w2, 128, 128);
  }
  const _Float16* p_de_w1 = pack(dec_edge.w1, 384, 128);
  const _Float16* p_de_w2 = pack(dec_edge.w2, 128, 128);
  const _Float16* p_dn_w1 = pack(dec_node.w1, 256, 128);
  const _Float16* p_dn_w2 = pack(dec_node.w2, 128, 128);
  const _Float16* p_f_w1  = pack(fin_w1, 128, 128);
  const _Float16* p_f_w2  = pack(fin_w2, 128, 64);

  // dynamic-LDS sizes (hidden 32KB + NSEG*32KB w1 + (DOUT/16)*4KB w2)
  constexpr int SH_F3 = 163840, SH_F2 = 131072, SH_F1 = 98304, SH_FF = 81920;
  constexpr int SH_EMB = 65536;
  auto setshm = [&](const void* f, int bytes) {
    (void)hipFuncSetAttribute(f, hipFuncAttributeMaxDynamicSharedMemorySize, bytes);
  };
  setshm(reinterpret_cast<const void*>(embed_k), SH_EMB);
  setshm(reinterpret_cast<const void*>(fused_block_k<3, 128, false, false>), SH_F3);
  setshm(reinterpret_cast<const void*>(fused_block_k<2, 128, true, false>), SH_F2);
  setshm(reinterpret_cast<const void*>(fused_block_k<1, 128, false, false>), SH_F1);
  setshm(reinterpret_cast<const void*>(fused_block_k<1, 64, false, true>), SH_FF);

  auto nb = [](int rows) { return dim3((rows + 127) / 128); };
  auto zero = [&](float* p, size_t n) {
    zero_k<<<dim3((unsigned)((n + 255) / 256)), dim3(256), 0, stream>>>(p, n);
  };
  auto scat = [&](const _Float16* e, const int* dsti, float* agg, int E) {
    scatter_add_k<<<dim3((E * 32 + 255) / 256), dim3(256), 0, stream>>>(e, dsti, agg, E);
  };

  // ---- embedder ----
  embed_k<<<nb(N_GRID), 256, SH_EMB, stream>>>(x_grid, 64, N_GRID, emb_grid.w1,
      emb_grid.b1, p_eg_w2, emb_grid.b2, emb_grid.g, emb_grid.beta, G);
  embed_k<<<nb(N_MESH), 256, SH_EMB, stream>>>(x_mesh, 3, N_MESH, emb_mesh.w1,
      emb_mesh.b1, p_em_w2, emb_mesh.b2, emb_mesh.g, emb_mesh.beta, Mn);
  embed_k<<<nb(E_M2M), 256, SH_EMB, stream>>>(x_m2m, 4, E_M2M, emb_m2m.w1,
      emb_m2m.b1, p_m2m_w2, emb_m2m.b2, emb_m2m.g, emb_m2m.beta, Em2m);
  embed_k<<<nb(E_G2M), 256, SH_EMB, stream>>>(x_g2m, 4, E_G2M, emb_g2m.w1,
      emb_g2m.b1, p_g2m_w2, emb_g2m.b2, emb_g2m.g, emb_g2m.beta, Eg2m);
  embed_k<<<nb(E_M2G), 256, SH_EMB, stream>>>(x_m2g, 4, E_M2G, emb_m2g.w1,
      emb_m2g.b1, p_m2g_w2, emb_m2g.b2, emb_m2g.g, emb_m2g.beta, Em2g);

  // ---- encoder (grid -> mesh) ----
  fused_block_k<3, 128, false, false><<<nb(E_G2M), 256, SH_F3, stream>>>(
      Eg2m, G, Mn, g2m_src, g2m_dst, E_G2M, p_ee_w1, enc_edge.b1, p_ee_w2,
      enc_edge.b2, enc_edge.g, enc_edge.beta, Eg2m, Eg2m);
  zero(AGG, (size_t)N_MESH * H);
  scat(Eg2m, g2m_dst, AGG, E_G2M);
  fused_block_k<2, 128, true, false><<<nb(N_MESH), 256, SH_F2, stream>>>(
      Mn, AGG, nullptr, nullptr, nullptr, N_MESH, p_en_w1, enc_node.b1,
      p_en_w2, enc_node.b2, enc_node.g, enc_node.beta, Mn, Mn);
  fused_block_k<1, 128, false, false><<<nb(N_GRID), 256, SH_F1, stream>>>(
      G, nullptr, nullptr, nullptr, nullptr, N_GRID, p_gg_w1, enc_grid.b1,
      p_gg_w2, enc_grid.b2, enc_grid.g, enc_grid.beta, G, G);

  // ---- processor (2 layers, mesh multigraph) ----
  for (int L = 0; L < 2; ++L) {
    fused_block_k<3, 128, false, false><<<nb(E_M2M), 256, SH_F3, stream>>>(
        Em2m, Mn, Mn, m2m_src, m2m_dst, E_M2M, p_pe_w1[L], proc_e[L].b1,
        p_pe_w2[L], proc_e[L].b2, proc_e[L].g, proc_e[L].beta, Em2m, Em2m);
    zero(AGG, (size_t)N_MESH * H);
    scat(Em2m, m2m_dst, AGG, E_M2M);
    fused_block_k<2, 128, true, false><<<nb(N_MESH), 256, SH_F2, stream>>>(
        Mn, AGG, nullptr, nullptr, nullptr, N_MESH, p_pn_w1[L], proc_n[L].b1,
        p_pn_w2[L], proc_n[L].b2, proc_n[L].g, proc_n[L].beta, Mn, Mn);
  }

  // ---- decoder (mesh -> grid) ----
  fused_block_k<3, 128, false, false><<<nb(E_M2G), 256, SH_F3, stream>>>(
      Em2g, Mn, G, m2g_src, m2g_dst, E_M2G, p_de_w1, dec_edge.b1, p_de_w2,
      dec_edge.b2, dec_edge.g, dec_edge.beta, Em2g, Em2g);
  zero(AGG, (size_t)N_GRID * H);
  scat(Em2g, m2g_dst, AGG, E_M2G);
  fused_block_k<2, 128, true, false><<<nb(N_GRID), 256, SH_F2, stream>>>(
      G, AGG, nullptr, nullptr, nullptr, N_GRID, p_dn_w1, dec_node.b1,
      p_dn_w2, dec_node.b2, dec_node.g, dec_node.beta, G, G);

  // ---- final prediction head (128 -> 128 -> 64, no LN, f32 out) ----
  fused_block_k<1, 64, false, true><<<nb(N_GRID), 256, SH_FF, stream>>>(
      G, nullptr, nullptr, nullptr, nullptr, N_GRID, p_f_w1, fin_b1, p_f_w2,
      fin_b2, nullptr, nullptr, nullptr, (float*)d_out);
}
